// CliffordRollingAttention_21225728376930
// MI455X (gfx1250) — matrix operand order repaired and checked by reading, not compiled, not run on hardware
//
#include <hip/hip_runtime.h>
#include <hip/hip_bf16.h>

// ---------------------------------------------------------------------------
// CliffordRollingAttention on gfx1250 (MI455X):
//   q/k/v/out projections  -> v_wmma_f32_16x16x32_f16 (fp32 accumulate),
//                             double-buffered LDS, one barrier per K-step
//   rmsnorm / softmax      -> fp32 VALU + wave32 shfl reductions
//   channel-rolled scores  -> LDS-staged q row, 5 rolled dot products
// ---------------------------------------------------------------------------

typedef _Float16 h4  __attribute__((ext_vector_type(4)));
typedef _Float16 h8  __attribute__((ext_vector_type(8)));
typedef _Float16 h16 __attribute__((ext_vector_type(16)));
typedef float    f8  __attribute__((ext_vector_type(8)));

#define HQ    16     // heads
#define DD    128    // head dim
#define LSEQ  2048   // sequence length
#define QD    2048   // query dim
#define INNER 2048   // H*D
#define TTOK  4096   // B*L tokens

// SEQ_SHIFTS from the reference (log-spaced, max_len=2048):
__constant__ int SHIFTS[16] = {0, 1, -1, 3, -3, 7, -7, 20, -20,
                               53, -53, 141, -141, 380, -380, 1024};

// ---------------------------------------------------------------------------
// fp32 -> fp16 conversion (vectorized x4)
// ---------------------------------------------------------------------------
__global__ __launch_bounds__(256)
void cvt_f32_to_f16x4(const float* __restrict__ in, _Float16* __restrict__ out, int n)
{
    const int n4 = n >> 2;
    for (int i = blockIdx.x * blockDim.x + threadIdx.x; i < n4;
         i += gridDim.x * blockDim.x) {
        float4 v = ((const float4*)in)[i];
        h4 o;
        o[0] = (_Float16)v.x; o[1] = (_Float16)v.y;
        o[2] = (_Float16)v.z; o[3] = (_Float16)v.w;
        ((h4*)out)[i] = o;
    }
}

// ---------------------------------------------------------------------------
// C[M,N] = A[M,K] * W[N,K]^T + bias   (A,W fp16 row-major, fp32 accumulate)
// Block: 256 thr = 8 waves; tile 128x128, BK=32; wave -> 2x4 WMMA subtiles.
// Double-buffered LDS: global loads for tile t+1 issue before computing tile
// t, stores land in the alternate buffer after compute; 1 barrier/iteration.
// ---------------------------------------------------------------------------
template <int OUT_HALF>
__global__ __launch_bounds__(256)
void gemm_wmma_f16(const _Float16* __restrict__ A,
                   const _Float16* __restrict__ W,
                   const float*    __restrict__ bias,
                   void* __restrict__ Cout,
                   int M, int N, int K)
{
    __shared__ _Float16 As[2][128][40];  // +8 halves pad: conflict-free b128
    __shared__ _Float16 Ws[2][128][40];

    const int tid  = threadIdx.x;
    const int wv   = tid >> 5;
    const int lane = tid & 31;
    const int r    = lane & 15;
    const int hh   = lane >> 4;       // which half of the wave
    const int wm   = wv >> 1;         // 0..3  (M direction)
    const int wn   = wv & 1;          // 0..1  (N direction)
    const int mBlk = blockIdx.y * 128;
    const int nBlk = blockIdx.x * 128;

    const int ldRow = tid >> 1;        // 0..127
    const int ldCol = (tid & 1) * 16;  // 0 or 16 (halves)

    f8 acc[2][4];
#pragma unroll
    for (int i = 0; i < 2; i++)
#pragma unroll
        for (int j = 0; j < 4; j++)
#pragma unroll
            for (int v = 0; v < 8; v++) acc[i][j][v] = 0.0f;

    const _Float16* Arow = A + (size_t)(mBlk + ldRow) * K + ldCol;
    const _Float16* Wrow = W + (size_t)(nBlk + ldRow) * K + ldCol;

    // prologue: stage tile 0 into buffer 0
    {
        h8 a0 = *(const h8*)(Arow);
        h8 a1 = *(const h8*)(Arow + 8);
        h8 w0 = *(const h8*)(Wrow);
        h8 w1 = *(const h8*)(Wrow + 8);
        *(h8*)(&As[0][ldRow][ldCol])     = a0;
        *(h8*)(&As[0][ldRow][ldCol + 8]) = a1;
        *(h8*)(&Ws[0][ldRow][ldCol])     = w0;
        *(h8*)(&Ws[0][ldRow][ldCol + 8]) = w1;
    }

    const int T = K >> 5;              // #K-steps
    for (int t = 0; t < T; ++t) {
        __syncthreads();               // buf[cur] fully staged, buf[nxt] free
        const int cur = t & 1;
        const int nxt = cur ^ 1;

        // issue global loads for tile t+1 (clamped on the tail: harmless
        // re-load of the last tile; the store lands in a dead buffer)
        const int kn = (t + 1 < T ? t + 1 : t) << 5;
        h8 a0 = *(const h8*)(Arow + kn);
        h8 a1 = *(const h8*)(Arow + kn + 8);
        h8 w0 = *(const h8*)(Wrow + kn);
        h8 w1 = *(const h8*)(Wrow + kn + 8);
        const int kp = (t + 2 < T ? t + 2 : T - 1) << 5;
        __builtin_prefetch(Arow + kp, 0, 3);   // global_prefetch_b8 (near)
        __builtin_prefetch(Wrow + kp, 0, 3);

        // gather all fragments for tile t, then chain the 8 WMMAs
        // A fragment: lanes 0-15 K{0..7,16..23}, lanes 16-31 K{8..15,24..31}
        h16 afr[2];
#pragma unroll
        for (int mi = 0; mi < 2; mi++) {
            const int row = wm * 32 + mi * 16 + r;
            h8 lo = *(const h8*)(&As[cur][row][hh * 8]);
            h8 hi = *(const h8*)(&As[cur][row][16 + hh * 8]);
#pragma unroll
            for (int e = 0; e < 8; e++) { afr[mi][e] = lo[e]; afr[mi][e + 8] = hi[e]; }
        }
        // B fragment: lanes 0-15 K{0..7,16..23}, lanes 16-31 K{8..15,24..31}
        h16 bfr[4];
#pragma unroll
        for (int ni = 0; ni < 4; ni++) {
            const int row = wn * 64 + ni * 16 + r;
            h8 lo = *(const h8*)(&Ws[cur][row][hh * 8]);
            h8 hi = *(const h8*)(&Ws[cur][row][16 + hh * 8]);
#pragma unroll
            for (int e = 0; e < 8; e++) { bfr[ni][e] = lo[e]; bfr[ni][e + 8] = hi[e]; }
        }
#pragma unroll
        for (int ni = 0; ni < 4; ni++)
#pragma unroll
            for (int mi = 0; mi < 2; mi++)
                acc[mi][ni] = __builtin_amdgcn_wmma_f32_16x16x32_f16(
                    false, afr[mi], false, bfr[ni], (short)0, acc[mi][ni],
                    false, false);

        // stage tile t+1 into the alternate buffer
        *(h8*)(&As[nxt][ldRow][ldCol])     = a0;
        *(h8*)(&As[nxt][ldRow][ldCol + 8]) = a1;
        *(h8*)(&Ws[nxt][ldRow][ldCol])     = w0;
        *(h8*)(&Ws[nxt][ldRow][ldCol + 8]) = w1;
    }

    // epilogue: C VGPR layout -> N = r, M = v + 8*hh (per 16x16 tile)
#pragma unroll
    for (int mi = 0; mi < 2; mi++) {
#pragma unroll
        for (int ni = 0; ni < 4; ni++) {
            const int col = nBlk + wn * 64 + ni * 16 + r;
            const float bb = bias[col];
#pragma unroll
            for (int v = 0; v < 8; v++) {
                const int row = mBlk + wm * 32 + mi * 16 + hh * 8 + v;
                const float val = acc[mi][ni][v] + bb;
                if (OUT_HALF)
                    ((_Float16*)Cout)[(size_t)row * N + col] = (_Float16)val;
                else
                    ((float*)Cout)[(size_t)row * N + col] = val;
            }
        }
    }
}

// ---------------------------------------------------------------------------
// RMSNorm over 2048-dim rows: out = x * rsqrt(mean(x^2)+eps) * g  (fp16 out)
// ---------------------------------------------------------------------------
__global__ __launch_bounds__(256)
void rmsnorm_f16(const float* __restrict__ in, const float* __restrict__ g,
                 _Float16* __restrict__ outp, int C)
{
    const int row = blockIdx.x;
    const float* rp = in + (size_t)row * C;
    float ss = 0.0f;
    for (int i = threadIdx.x; i < C; i += 256) { float v = rp[i]; ss += v * v; }
#pragma unroll
    for (int off = 16; off; off >>= 1) ss += __shfl_xor(ss, off);
    __shared__ float wsum[8];
    if ((threadIdx.x & 31) == 0) wsum[threadIdx.x >> 5] = ss;
    __syncthreads();
    float tot = 0.0f;
#pragma unroll
    for (int i = 0; i < 8; i++) tot += wsum[i];
    const float rs = rsqrtf(tot / (float)C + 1e-6f);
    for (int i = threadIdx.x; i < C; i += 256)
        outp[(size_t)row * C + i] = (_Float16)(rp[i] * rs * g[i]);
}

// ---------------------------------------------------------------------------
// Rolling attention: one wave per (b,l,h).
// scores[s,c] = sum_d q[(d-c)&127] * k_shift_s[d];  mixed = scale * scores@wmix
// out = softmax_s(mixed) @ v_shift
// ---------------------------------------------------------------------------
__global__ __launch_bounds__(256)
void attention_kernel(const _Float16* __restrict__ qh,
                      const _Float16* __restrict__ kh,
                      const _Float16* __restrict__ vh,
                      const float*    __restrict__ wmix,
                      _Float16* __restrict__ ah)
{
    __shared__ float qsm[8][DD];
    __shared__ float mix[8][16];

    const int w    = threadIdx.x >> 5;
    const int lane = threadIdx.x & 31;
    const long g   = (long)blockIdx.x * 8 + w;
    const int hid  = (int)(g & (HQ - 1));
    const int l    = (int)((g >> 4) & (LSEQ - 1));
    const int b    = (int)(g >> 15);                   // 16*2048 = 2^15

    const size_t qbase = ((((size_t)b * LSEQ + l) * HQ + hid) << 7);

    h4 qv = *(const h4*)(qh + qbase + lane * 4);
#pragma unroll
    for (int i = 0; i < 4; i++) qsm[w][lane * 4 + i] = (float)qv[i];

    const float w0 = wmix[0], w1 = wmix[1], w2 = wmix[2], w3 = wmix[3], w4 = wmix[4];
    const float scale = 0.08838834764831845f;          // D^-0.5

    for (int s = 0; s < 16; s++) {
        const int ls = (l - SHIFTS[s] + LSEQ) & (LSEQ - 1);
        const size_t kb = ((((size_t)b * LSEQ + ls) * HQ + hid) << 7);
        h4 kv = *(const h4*)(kh + kb + lane * 4);
        float p0 = 0, p1 = 0, p2 = 0, p3 = 0, p4 = 0;
#pragma unroll
        for (int i = 0; i < 4; i++) {
            const int d = lane * 4 + i;
            const float kk = (float)kv[i];
            p0 += kk * qsm[w][d];
            p1 += kk * qsm[w][(d - 1) & (DD - 1)];
            p2 += kk * qsm[w][(d - 2) & (DD - 1)];
            p3 += kk * qsm[w][(d - 4) & (DD - 1)];
            p4 += kk * qsm[w][(d - 8) & (DD - 1)];
        }
#pragma unroll
        for (int off = 16; off; off >>= 1) {
            p0 += __shfl_xor(p0, off); p1 += __shfl_xor(p1, off);
            p2 += __shfl_xor(p2, off); p3 += __shfl_xor(p3, off);
            p4 += __shfl_xor(p4, off);
        }
        if (lane == 0)
            mix[w][s] = scale * (w0 * p0 + w1 * p1 + w2 * p2 + w3 * p3 + w4 * p4);
    }

    // softmax over the 16 shifts (per-wave data; LDS ops are in-order per wave)
    float mx = -1e30f;
#pragma unroll
    for (int s = 0; s < 16; s++) mx = fmaxf(mx, mix[w][s]);
    float e[16], sum = 0.0f;
#pragma unroll
    for (int s = 0; s < 16; s++) { e[s] = __expf(mix[w][s] - mx); sum += e[s]; }
    const float inv = 1.0f / sum;

    float o0 = 0, o1 = 0, o2 = 0, o3 = 0;
    for (int s = 0; s < 16; s++) {
        const int ls = (l - SHIFTS[s] + LSEQ) & (LSEQ - 1);
        const size_t vb = ((((size_t)b * LSEQ + ls) * HQ + hid) << 7);
        h4 vv = *(const h4*)(vh + vb + lane * 4);
        const float wgt = e[s] * inv;
        o0 += wgt * (float)vv[0]; o1 += wgt * (float)vv[1];
        o2 += wgt * (float)vv[2]; o3 += wgt * (float)vv[3];
    }
    h4 ov;
    ov[0] = (_Float16)o0; ov[1] = (_Float16)o1;
    ov[2] = (_Float16)o2; ov[3] = (_Float16)o3;
    *(h4*)(ah + qbase + lane * 4) = ov;
}

// ---------------------------------------------------------------------------
extern "C" void kernel_launch(void* const* d_in, const int* in_sizes, int n_in,
                              void* d_out, int out_size, void* d_ws, size_t ws_size,
                              hipStream_t stream)
{
    const float* x    = (const float*)d_in[0];
    const float* wq   = (const float*)d_in[1];
    const float* bq   = (const float*)d_in[2];
    const float* wk   = (const float*)d_in[3];
    const float* bk   = (const float*)d_in[4];
    const float* wv   = (const float*)d_in[5];
    const float* bv   = (const float*)d_in[6];
    const float* gq   = (const float*)d_in[7];
    const float* gk   = (const float*)d_in[8];
    const float* wmix = (const float*)d_in[9];
    const float* wo   = (const float*)d_in[10];
    const float* bo   = (const float*)d_in[11];
    float* out = (float*)d_out;

    char* base = (char*)d_ws;
    size_t off = 0;
    auto alloc = [&](size_t bytes) -> void* {
        void* p = base + off;
        off += (bytes + 255) & ~(size_t)255;
        return p;
    };
    _Float16* xh   = (_Float16*)alloc((size_t)TTOK  * QD * 2);
    _Float16* wqh  = (_Float16*)alloc((size_t)INNER * QD * 2);
    _Float16* wkh  = (_Float16*)alloc((size_t)INNER * QD * 2);
    _Float16* wvh  = (_Float16*)alloc((size_t)INNER * QD * 2);
    _Float16* woh  = (_Float16*)alloc((size_t)QD * INNER * 2);
    float*    qraw = (float*)   alloc((size_t)TTOK * INNER * 4);
    float*    kraw = (float*)   alloc((size_t)TTOK * INNER * 4);
    _Float16* qh   = (_Float16*)alloc((size_t)TTOK * INNER * 2);
    _Float16* kh   = (_Float16*)alloc((size_t)TTOK * INNER * 2);
    _Float16* vh   = (_Float16*)alloc((size_t)TTOK * INNER * 2);
    _Float16* ah   = (_Float16*)alloc((size_t)TTOK * INNER * 2);

    // fp32 -> fp16 staging
    cvt_f32_to_f16x4<<<2048, 256, 0, stream>>>(x,  xh,  TTOK  * QD);
    cvt_f32_to_f16x4<<<1024, 256, 0, stream>>>(wq, wqh, INNER * QD);
    cvt_f32_to_f16x4<<<1024, 256, 0, stream>>>(wk, wkh, INNER * QD);
    cvt_f32_to_f16x4<<<1024, 256, 0, stream>>>(wv, wvh, INNER * QD);
    cvt_f32_to_f16x4<<<1024, 256, 0, stream>>>(wo, woh, QD * INNER);

    // projections
    dim3 gp(INNER / 128, TTOK / 128);
    gemm_wmma_f16<0><<<gp, 256, 0, stream>>>(xh, wqh, bq, qraw, TTOK, INNER, QD);
    gemm_wmma_f16<0><<<gp, 256, 0, stream>>>(xh, wkh, bk, kraw, TTOK, INNER, QD);
    gemm_wmma_f16<1><<<gp, 256, 0, stream>>>(xh, wvh, bv, vh,   TTOK, INNER, QD);

    // rmsnorm q, k
    rmsnorm_f16<<<TTOK, 256, 0, stream>>>(qraw, gq, qh, INNER);
    rmsnorm_f16<<<TTOK, 256, 0, stream>>>(kraw, gk, kh, INNER);

    // rolling attention: one wave per (b,l,h) -> TTOK*HQ waves, 8 per block
    attention_kernel<<<(TTOK * HQ) / 8, 256, 0, stream>>>(qh, kh, vh, wmix, ah);

    // output projection -> fp32 d_out
    dim3 go(QD / 128, TTOK / 128);
    gemm_wmma_f16<0><<<go, 256, 0, stream>>>(ah, woh, bo, out, TTOK, QD, INNER);
}
